// FusedMoE_85057532330524
// MI455X (gfx1250) — matrix operand order repaired and checked by reading, not compiled, not run on hardware
//
#include <hip/hip_runtime.h>
#include <cstdint>
#include <cstddef>

// ---------------------------------------------------------------------------
// Problem constants (from reference): T=4096 H=1024 I=2048 E=8 TOPK=2
// ---------------------------------------------------------------------------
#define T_TOK   4096
#define HDIM    1024
#define IDIM    2048
#define NEXP    8
#define TOPK    2
#define NASSIGN (T_TOK * TOPK)   // 8192 assignment slots

// GEMM tiling: 256 threads = 8 waves (wave32). Block tile 64(M) x 128(N),
// K-step 64 (two 16x16x32 sub-steps). Wave grid 2(m) x 4(n): each wave owns a
// 32x32 output tile = 2 A-frags x 2 B-frags -> 4 wmma per matrix per substep,
// with every fragment reused twice (one dscnt wait per substep).
#define MT 64
#define NT 128
#define KT 64
#define LROW 72   // 64 bf16 + 8 pad -> 144B row: 16B aligned, bank-staggered

typedef __attribute__((ext_vector_type(16))) __bf16   v16bf;
typedef __attribute__((ext_vector_type(8)))  __bf16   v8bf;
typedef __attribute__((ext_vector_type(8)))  float    v8f;
typedef __attribute__((ext_vector_type(4)))  float    f32x4;
typedef __attribute__((ext_vector_type(4)))  uint32_t u32x4;

#define WMMA_BF16(A, B, C) \
  __builtin_amdgcn_wmma_f32_16x16x32_bf16(false, (A), false, (B), (short)0, (C), false, false)

union AFrag { v16bf v; v8bf h[2]; };

// round-to-nearest-even f32->bf16, two at a time, packed into one dword
__device__ __forceinline__ uint32_t pack2bf(float lo, float hi) {
  uint32_t ul = __builtin_bit_cast(uint32_t, lo);
  uint32_t uh = __builtin_bit_cast(uint32_t, hi);
  ul += 0x7FFFu + ((ul >> 16) & 1u);
  uh += 0x7FFFu + ((uh >> 16) & 1u);
  return (ul >> 16) | (uh & 0xFFFF0000u);
}

// 8 floats (two f32x4) -> v8bf, fully packed
__device__ __forceinline__ v8bf cvt8(f32x4 a, f32x4 b) {
  u32x4 r;
  r.x = pack2bf(a.x, a.y);
  r.y = pack2bf(a.z, a.w);
  r.z = pack2bf(b.x, b.y);
  r.w = pack2bf(b.z, b.w);
  return __builtin_bit_cast(v8bf, r);
}

__device__ __forceinline__ __bf16 f2bf(float f) {
  uint32_t u = __builtin_bit_cast(uint32_t, f);
  uint32_t r = u + 0x7FFFu + ((u >> 16) & 1u);
  return __builtin_bit_cast(__bf16, (uint16_t)(r >> 16));
}

// ---------------------------------------------------------------------------
// 1) init: zero per-expert counters
// ---------------------------------------------------------------------------
__global__ void moe_init_kernel(int* counts) {
  if (threadIdx.x < NEXP) counts[threadIdx.x] = 0;
}

// ---------------------------------------------------------------------------
// 2) router: softmax + top2 + renormalize; histogram counts
// ---------------------------------------------------------------------------
__global__ void moe_router_kernel(const float* __restrict__ logits,
                                  int* __restrict__ topk_id,
                                  float* __restrict__ topk_w,
                                  int* __restrict__ counts) {
  int t = blockIdx.x * blockDim.x + threadIdx.x;
  if (t >= T_TOK) return;
  float p[NEXP];
  float mx = -1e30f;
  #pragma unroll
  for (int e = 0; e < NEXP; ++e) { p[e] = logits[t * NEXP + e]; mx = fmaxf(mx, p[e]); }
  float s = 0.f;
  #pragma unroll
  for (int e = 0; e < NEXP; ++e) { p[e] = __expf(p[e] - mx); s += p[e]; }
  float inv = 1.0f / s;
  #pragma unroll
  for (int e = 0; e < NEXP; ++e) p[e] *= inv;
  int e0 = 0; float b0 = p[0];
  #pragma unroll
  for (int e = 1; e < NEXP; ++e) if (p[e] > b0) { b0 = p[e]; e0 = e; }
  int e1 = (e0 == 0) ? 1 : 0; float b1 = p[e1];
  #pragma unroll
  for (int e = 0; e < NEXP; ++e) if (e != e0 && p[e] > b1) { b1 = p[e]; e1 = e; }
  float rs = 1.0f / (b0 + b1);                 // RENORMALIZE
  topk_id[t * 2 + 0] = e0; topk_w[t * 2 + 0] = b0 * rs;
  topk_id[t * 2 + 1] = e1; topk_w[t * 2 + 1] = b1 * rs;
  atomicAdd(&counts[e0], 1);
  atomicAdd(&counts[e1], 1);
}

// ---------------------------------------------------------------------------
// 3) scan: 8-entry exclusive scan -> offsets + cursors
// ---------------------------------------------------------------------------
__global__ void moe_scan_kernel(const int* __restrict__ counts,
                                int* __restrict__ offs,
                                int* __restrict__ cursors) {
  if (threadIdx.x == 0) {
    int acc = 0;
    for (int e = 0; e < NEXP; ++e) { offs[e] = acc; cursors[e] = acc; acc += counts[e]; }
    offs[NEXP] = acc;
  }
}

// ---------------------------------------------------------------------------
// 4) scatter: expert-sorted slot lists
// ---------------------------------------------------------------------------
__global__ void moe_scatter_kernel(const int* __restrict__ topk_id,
                                   const float* __restrict__ topk_w,
                                   int* __restrict__ cursors,
                                   int* __restrict__ perm_tok,
                                   float* __restrict__ perm_w,
                                   int* __restrict__ slot_of) {
  int t = blockIdx.x * blockDim.x + threadIdx.x;
  if (t >= T_TOK) return;
  #pragma unroll
  for (int k = 0; k < TOPK; ++k) {
    int e = topk_id[t * 2 + k];
    int pos = atomicAdd(&cursors[e], 1);
    perm_tok[pos] = t;
    perm_w[pos]   = topk_w[t * 2 + k];
    slot_of[t * 2 + k] = pos;
  }
}

// ---------------------------------------------------------------------------
// 5) GEMM1: h = silu(x@w1^T) * (x@w3^T)   per expert group  -> hbuf (bf16)
//    grid: (64 M-tiles max, IDIM/NT=16, NEXP)
// ---------------------------------------------------------------------------
__global__ __launch_bounds__(256)
void moe_gemm1_kernel(const float* __restrict__ x,
                      const float* __restrict__ w13,
                      const int* __restrict__ counts,
                      const int* __restrict__ offs,
                      const int* __restrict__ perm_tok,
                      __bf16* __restrict__ hbuf) {
  const int e   = blockIdx.z;
  const int cnt = counts[e];
  const int m0  = blockIdx.x * MT;
  if (m0 >= cnt) return;
  const int n0   = blockIdx.y * NT;      // over IDIM
  const int base = offs[e];

  __shared__ __align__(16) __bf16 sA [MT * LROW];
  __shared__ __align__(16) __bf16 sB1[NT * LROW];
  __shared__ __align__(16) __bf16 sB3[NT * LROW];

  const int tid  = threadIdx.x;
  const int lane = tid & 31;
  const int wave = tid >> 5;
  const int wm   = wave & 1;             // 0..1 -> 32-row chunk
  const int wn   = wave >> 1;            // 0..3 -> 32-col chunk
  const int kh   = lane >> 4, ml = lane & 15;

  // staging roles
  const int ar = tid >> 2, aq = tid & 3;         // A: 64 rows x 4 chunks(16 floats)
  const bool avalid = (m0 + ar) < cnt;
  const int  tok    = avalid ? perm_tok[base + m0 + ar] : 0;
  const float* xrow = x + (size_t)tok * HDIM;
  const int bn = tid >> 1, bh = tid & 1;         // B: 128 rows x 2 chunks(32 floats)
  const float* w1row = w13 + ((size_t)e * (2 * IDIM) + (n0 + bn)) * HDIM;
  const float* w3row = w13 + ((size_t)e * (2 * IDIM) + IDIM + (n0 + bn)) * HDIM;

  const v8bf z8 = __builtin_bit_cast(v8bf, (u32x4)(0u));
  v8f accG[4] = {}; v8f accU[4] = {};   // [ai*2+bi]

  for (int k0 = 0; k0 < HDIM; k0 += KT) {
    __syncthreads();
    { // stage A: two global_load_b128 pairs + packed cvt (branch hoisted)
      v8bf lo = z8, hi = z8;
      if (avalid) {
        const f32x4* p = (const f32x4*)(xrow + k0 + aq * 16);
        lo = cvt8(p[0], p[1]);
        hi = cvt8(p[2], p[3]);
      }
      *(v8bf*)&sA[ar * LROW + aq * 16 + 0] = lo;
      *(v8bf*)&sA[ar * LROW + aq * 16 + 8] = hi;
    }
    { // stage B1/B3: vector loads + packed cvt, prefetch next K-step
      const f32x4* p1 = (const f32x4*)(w1row + k0 + bh * 32);
      const f32x4* p3 = (const f32x4*)(w3row + k0 + bh * 32);
      if (k0 + KT < HDIM) {
        __builtin_prefetch(p1 + 16, 0, 1);   // +64 floats
        __builtin_prefetch(p3 + 16, 0, 1);
      }
      #pragma unroll
      for (int q = 0; q < 4; ++q) {
        *(v8bf*)&sB1[bn * LROW + bh * 32 + q * 8] = cvt8(p1[2 * q], p1[2 * q + 1]);
        *(v8bf*)&sB3[bn * LROW + bh * 32 + q * 8] = cvt8(p3[2 * q], p3[2 * q + 1]);
      }
    }
    __syncthreads();

    #pragma unroll
    for (int kk = 0; kk < KT; kk += 32) {
      // preload ALL fragments for this substep, then 8 back-to-back wmma
      AFrag a0, a1;
      const int ab0 = (wm * 32 + ml) * LROW + kk + kh * 16;
      const int ab1 = ab0 + 16 * LROW;
      a0.h[0] = *(const v8bf*)&sA[ab0];
      a0.h[1] = *(const v8bf*)&sA[ab0 + 8];
      a1.h[0] = *(const v8bf*)&sA[ab1];
      a1.h[1] = *(const v8bf*)&sA[ab1 + 8];
      const int nb0 = (wn * 32 + ml) * LROW + kk + kh * 16;
      const int nb1 = nb0 + 16 * LROW;
      v16bf b1x = *(const v16bf*)&sB1[nb0];
      v16bf b1y = *(const v16bf*)&sB1[nb1];
      v16bf b3x = *(const v16bf*)&sB3[nb0];
      v16bf b3y = *(const v16bf*)&sB3[nb1];
      accG[0] = WMMA_BF16(a0.v, b1x, accG[0]);
      accG[1] = WMMA_BF16(a0.v, b1y, accG[1]);
      accG[2] = WMMA_BF16(a1.v, b1x, accG[2]);
      accG[3] = WMMA_BF16(a1.v, b1y, accG[3]);
      accU[0] = WMMA_BF16(a0.v, b3x, accU[0]);
      accU[1] = WMMA_BF16(a0.v, b3y, accU[1]);
      accU[2] = WMMA_BF16(a1.v, b3x, accU[2]);
      accU[3] = WMMA_BF16(a1.v, b3y, accU[3]);
    }
  }

  // epilogue: silu(gate)*up -> hbuf (C layout: m = i + 8*(lane/16), n = lane%16)
  const int nl = lane & 15, half = lane >> 4;
  #pragma unroll
  for (int ai = 0; ai < 2; ++ai) {
    #pragma unroll
    for (int bi = 0; bi < 2; ++bi) {
      const int idx = ai * 2 + bi;
      #pragma unroll
      for (int i = 0; i < 8; ++i) {
        int m = wm * 32 + ai * 16 + i + half * 8;
        if (m0 + m < cnt) {
          float g = accG[idx][i], u = accU[idx][i];
          float hv = (g / (1.0f + __expf(-g))) * u;
          int col = n0 + wn * 32 + bi * 16 + nl;
          hbuf[(size_t)(base + m0 + m) * IDIM + col] = f2bf(hv);
        }
      }
    }
  }
}

// ---------------------------------------------------------------------------
// 6) GEMM2: y = coeff * (h @ w2^T)  -> ybuf (fp32, one row per slot)
//    grid: (64 M-tiles max, HDIM/NT=8, NEXP)
// ---------------------------------------------------------------------------
__global__ __launch_bounds__(256)
void moe_gemm2_kernel(const __bf16* __restrict__ hbuf,
                      const float* __restrict__ w2,
                      const int* __restrict__ counts,
                      const int* __restrict__ offs,
                      const float* __restrict__ perm_w,
                      float* __restrict__ ybuf) {
  const int e   = blockIdx.z;
  const int cnt = counts[e];
  const int m0  = blockIdx.x * MT;
  if (m0 >= cnt) return;
  const int n0   = blockIdx.y * NT;      // over HDIM
  const int base = offs[e];

  __shared__ __align__(16) __bf16 sA[MT * LROW];
  __shared__ __align__(16) __bf16 sB[NT * LROW];

  const int tid  = threadIdx.x;
  const int lane = tid & 31;
  const int wave = tid >> 5;
  const int wm   = wave & 1;
  const int wn   = wave >> 1;
  const int kh   = lane >> 4, ml = lane & 15;

  const int ar = tid >> 2, aq = tid & 3;          // A: 64 rows x 4 chunks(16 bf16)
  const bool avalid = (m0 + ar) < cnt;
  const __bf16* hrow = hbuf + (size_t)(base + m0 + ar) * IDIM;
  const int bn = tid >> 1, bh = tid & 1;          // B: 128 rows x 2 chunks(32 floats)
  const float* w2row = w2 + ((size_t)e * HDIM + (n0 + bn)) * IDIM;

  const v8bf z8 = __builtin_bit_cast(v8bf, (u32x4)(0u));
  v8f acc[4] = {};   // [ai*2+bi]

  for (int k0 = 0; k0 < IDIM; k0 += KT) {
    __syncthreads();
    { // stage A (already bf16): two 16B copies
      v8bf lo = z8, hi = z8;
      if (avalid) {
        const v8bf* p = (const v8bf*)(hrow + k0 + aq * 16);
        lo = p[0]; hi = p[1];
      }
      *(v8bf*)&sA[ar * LROW + aq * 16 + 0] = lo;
      *(v8bf*)&sA[ar * LROW + aq * 16 + 8] = hi;
    }
    { // stage B
      const f32x4* p = (const f32x4*)(w2row + k0 + bh * 32);
      if (k0 + KT < IDIM) __builtin_prefetch(p + 16, 0, 1);
      #pragma unroll
      for (int q = 0; q < 4; ++q)
        *(v8bf*)&sB[bn * LROW + bh * 32 + q * 8] = cvt8(p[2 * q], p[2 * q + 1]);
    }
    __syncthreads();

    #pragma unroll
    for (int kk = 0; kk < KT; kk += 32) {
      AFrag a0, a1;
      const int ab0 = (wm * 32 + ml) * LROW + kk + kh * 16;
      const int ab1 = ab0 + 16 * LROW;
      a0.h[0] = *(const v8bf*)&sA[ab0];
      a0.h[1] = *(const v8bf*)&sA[ab0 + 8];
      a1.h[0] = *(const v8bf*)&sA[ab1];
      a1.h[1] = *(const v8bf*)&sA[ab1 + 8];
      const int nb0 = (wn * 32 + ml) * LROW + kk + kh * 16;
      const int nb1 = nb0 + 16 * LROW;
      v16bf bx = *(const v16bf*)&sB[nb0];
      v16bf by = *(const v16bf*)&sB[nb1];
      acc[0] = WMMA_BF16(a0.v, bx, acc[0]);
      acc[1] = WMMA_BF16(a0.v, by, acc[1]);
      acc[2] = WMMA_BF16(a1.v, bx, acc[2]);
      acc[3] = WMMA_BF16(a1.v, by, acc[3]);
    }
  }

  // epilogue: scale by routing weight, store fp32 rows
  const int nl = lane & 15, half = lane >> 4;
  #pragma unroll
  for (int ai = 0; ai < 2; ++ai) {
    float rw[8]; bool mv[8];
    #pragma unroll
    for (int i = 0; i < 8; ++i) {
      int m = wm * 32 + ai * 16 + i + half * 8;
      mv[i] = (m0 + m) < cnt;
      rw[i] = mv[i] ? perm_w[base + m0 + m] : 0.0f;
    }
    #pragma unroll
    for (int bi = 0; bi < 2; ++bi) {
      const int idx = ai * 2 + bi;
      #pragma unroll
      for (int i = 0; i < 8; ++i) {
        if (mv[i]) {
          int m   = wm * 32 + ai * 16 + i + half * 8;
          int col = n0 + wn * 32 + bi * 16 + nl;
          ybuf[(size_t)(base + m0 + m) * HDIM + col] = rw[i] * acc[idx][i];
        }
      }
    }
  }
}

// ---------------------------------------------------------------------------
// 7) combine: out[t] = ybuf[slot0] + ybuf[slot1]   (deterministic, no atomics)
// ---------------------------------------------------------------------------
__global__ void moe_combine_kernel(const float* __restrict__ ybuf,
                                   const int* __restrict__ slot_of,
                                   float* __restrict__ out) {
  int t = blockIdx.x;
  int s0 = slot_of[t * 2 + 0];
  int s1 = slot_of[t * 2 + 1];
  const f32x4* y0 = (const f32x4*)(ybuf + (size_t)s0 * HDIM);
  const f32x4* y1 = (const f32x4*)(ybuf + (size_t)s1 * HDIM);
  f32x4* o = (f32x4*)(out + (size_t)t * HDIM);
  for (int c = threadIdx.x; c < HDIM / 4; c += blockDim.x)
    o[c] = y0[c] + y1[c];
}

// ---------------------------------------------------------------------------
// launch
// ---------------------------------------------------------------------------
extern "C" void kernel_launch(void* const* d_in, const int* in_sizes, int n_in,
                              void* d_out, int out_size, void* d_ws, size_t ws_size,
                              hipStream_t stream) {
  const float* x      = (const float*)d_in[0];
  const float* logits = (const float*)d_in[1];
  const float* w13    = (const float*)d_in[2];
  const float* w2     = (const float*)d_in[3];
  float* out          = (float*)d_out;

  // workspace layout
  char* ws = (char*)d_ws;
  int*    topk_id  = (int*)(ws);                          //  32 KB
  float*  topk_w   = (float*)(ws + (32 << 10));           //  32 KB
  int*    counts   = (int*)(ws + (64 << 10));             //  8
  int*    cursors  = (int*)(ws + (64 << 10) + 64);        //  8
  int*    offs     = (int*)(ws + (64 << 10) + 128);       //  9
  int*    perm_tok = (int*)(ws + (64 << 10) + 256);       //  32 KB
  float*  perm_w   = (float*)(ws + (96 << 10) + 256);     //  32 KB
  int*    slot_of  = (int*)(ws + (128 << 10) + 256);      //  32 KB
  __bf16* hbuf     = (__bf16*)(ws + (1ull << 20));                             // 32 MB
  float*  ybuf     = (float*)(ws + (1ull << 20) + (size_t)NASSIGN * IDIM * 2); // 32 MB

  moe_init_kernel<<<1, 32, 0, stream>>>(counts);
  moe_router_kernel<<<T_TOK / 256, 256, 0, stream>>>(logits, topk_id, topk_w, counts);
  moe_scan_kernel<<<1, 32, 0, stream>>>(counts, offs, cursors);
  moe_scatter_kernel<<<T_TOK / 256, 256, 0, stream>>>(topk_id, topk_w, cursors,
                                                      perm_tok, perm_w, slot_of);
  dim3 g1(T_TOK / MT, IDIM / NT, NEXP);   // (64, 16, 8) — early-exit past counts
  moe_gemm1_kernel<<<g1, 256, 0, stream>>>(x, w13, counts, offs, perm_tok, hbuf);
  dim3 g2(T_TOK / MT, HDIM / NT, NEXP);   // (64, 8, 8)
  moe_gemm2_kernel<<<g2, 256, 0, stream>>>(hbuf, w2, counts, offs, perm_w, ybuf);
  moe_combine_kernel<<<T_TOK, 256, 0, stream>>>(ybuf, slot_of, out);
}
